// SelfHealingQuantumCell_18279380812191
// MI455X (gfx1250) — matrix operand order repaired and checked by reading, not compiled, not run on hardware
//
#include <hip/hip_runtime.h>

// ---- CDNA5 (gfx1250) types ------------------------------------------------
typedef __attribute__((ext_vector_type(16))) __bf16         v16bf;
typedef __attribute__((ext_vector_type(16))) unsigned short v16u;
typedef __attribute__((ext_vector_type(8)))  float          v8f;
typedef __attribute__((ext_vector_type(16))) float          v16f;
typedef __attribute__((ext_vector_type(4)))  float          v4f;

union FragAB { v16bf b; v16u u; };   // 32B WMMA A/B operand (8 VGPRs)

// Hardware tanh (V_TANH_F32 on gfx1250) instead of the branchy libm expansion.
#if __has_builtin(__builtin_amdgcn_tanhf)
__device__ __forceinline__ float fast_tanh(float v) { return __builtin_amdgcn_tanhf(v); }
#elif __has_builtin(__builtin_amdgcn_tanh_f32)
__device__ __forceinline__ float fast_tanh(float v) { return __builtin_amdgcn_tanh_f32(v); }
#else
__device__ __forceinline__ float fast_tanh(float v) { return tanhf(v); }
#endif

// Native ~1ulp transcendentals: avoid the v_div_scale/v_div_fmas IEEE chains.
__device__ __forceinline__ float fast_rcp(float v)  { return __builtin_amdgcn_rcpf(v); }
__device__ __forceinline__ float fast_sqrt(float v) { return __builtin_amdgcn_sqrtf(v); }

// fp32 -> bf16 (RNE) via the hardware convert path.
__device__ __forceinline__ __bf16 f2bf(float f) { return (__bf16)f; }

// ---------------------------------------------------------------------------
// Prep: weights -> bf16 in per-lane WMMA B-fragment order in d_ws.
// B layout (16x16x32 bf16, wave32): lane L holds N = L&15;
//   slot e (0..15) holds K = e<8 ? 8*(L>>4)+e : 16+8*(L>>4)+(e-8).
// Flat ws index: (((s*2 + t)*32 + L)*16 + e), t = k-half.
// recW: B[K=h][N=k] = recurrent_weights[h][k][s]   (h,k in 0..31)
// qwW : B[K=i][N=k] = quantum_weights[i][k][s] for i<8 else 0 (K zero-padded)
// ---------------------------------------------------------------------------
__global__ __launch_bounds__(256) void prep_weights_kernel(
    const float* __restrict__ qw, const float* __restrict__ rw,
    __bf16* __restrict__ wsRec, __bf16* __restrict__ wsQw)
{
  int id = blockIdx.x * 256 + threadIdx.x;
  if (id >= 16 * 2 * 32 * 16) return;
  int e = id & 15;
  int L = (id >> 4) & 31;
  int t = (id >> 9) & 1;
  int s = id >> 10;
  int K = ((L >> 4) << 3) + (e & 7) + ((e >> 3) << 4);
  int N = (t << 4) + (L & 15);
  wsRec[id] = f2bf(rw[(K * 32 + N) * 16 + s]);
  wsQw[id]  = (K < 8) ? f2bf(qw[(K * 32 + N) * 16 + s]) : (__bf16)0.0f;
}

// ---------------------------------------------------------------------------
// Main: one wave (32 threads) owns a 16-row batch tile.
// ---------------------------------------------------------------------------
__global__ __launch_bounds__(32) void shqc_kernel(
    const float* __restrict__ x,    const float* __restrict__ hq,
    const float* __restrict__ btau, const float* __restrict__ syn,
    const float* __restrict__ corr,
    const __bf16* __restrict__ wsRec,
    const __bf16* __restrict__ wsQw,
    float* __restrict__ out, float* __restrict__ qnorm)
{
  __shared__ __align__(32) __bf16 hqA[16 * 32 * 16];  // [s][lane][slot] bf16 A-frags (16KB)
  __shared__ __align__(16) float msyn[16 * 32 * 4];   // [b_local][h][c]          (8KB)
  __shared__ __align__(64) float stab[16 * 32 * 16];  // [b_local][k][s]          (32KB)
  __shared__ float synL[64];
  __shared__ float corrL[64];

  const int  lane = threadIdx.x;
  const int  half = lane >> 4;
  const int  bl   = lane & 15;
  const long b0   = (long)blockIdx.x << 4;
  const long bg   = b0 + bl;

  // Warm L2 for the shared bf16 weight-fragment tables (global_prefetch_b8).
  __builtin_prefetch(&wsRec[lane << 9], 0, 1);
  __builtin_prefetch(&wsQw[lane << 9], 0, 1);

  synL[lane]       = syn[lane];
  synL[lane + 32]  = syn[lane + 32];
  corrL[lane]      = corr[lane];
  corrL[lane + 32] = corr[lane + 32];
  __syncthreads();

  // ---- Phase 1: syndrome + threshold mask + correction, stage bf16 A-frags
  {
    const int hbase = half << 4;
    for (int hh = 0; hh < 16; ++hh) {
      const int h = hbase + hh;
      v16f row = *(const v16f*)(hq + ((bg * 32 + h) << 4));
      float d0 = 0.f, d1 = 0.f, d2 = 0.f, d3 = 0.f;
      #pragma unroll
      for (int s = 0; s < 16; ++s) {
        const float v = row[s];
        d0 += v * synL[s];
        d1 += v * synL[16 + s];
        d2 += v * synL[32 + s];
        d3 += v * synL[48 + s];
      }
      const float ms0 = (fabsf(d0) > 0.01f) ? d0 : 0.f;
      const float ms1 = (fabsf(d1) > 0.01f) ? d1 : 0.f;
      const float ms2 = (fabsf(d2) > 0.01f) ? d2 : 0.f;
      const float ms3 = (fabsf(d3) > 0.01f) ? d3 : 0.f;
      v4f msv = {ms0, ms1, ms2, ms3};
      *(v4f*)&msyn[(bl * 32 + h) << 2] = msv;
      #pragma unroll
      for (int s = 0; s < 16; ++s)
        row[s] -= ms0 * corrL[s] + ms1 * corrL[16 + s] +
                  ms2 * corrL[32 + s] + ms3 * corrL[48 + s];
      // A-fragment slot mapping: K(e,half') = e<8 ? 8*half'+e : 16+8*half'+(e-8)
      const int halfp = (h < 16) ? (h >> 3) : ((h - 16) >> 3);
      const int ep    = (h < 16) ? (h & 7)  : (8 + ((h - 16) & 7));
      const int lanep = (halfp << 4) + bl;
      #pragma unroll
      for (int s = 0; s < 16; ++s)
        hqA[((s * 32 + lanep) << 4) + ep] = f2bf(row[s]);
    }
  }
  __syncthreads();

  // ---- Phase 2: WMMA GEMMs + tanh/Euler + entanglement (lagged by one s)
  FragAB ax;                               // x A-frag, K padded 8->32 with zeros
  #pragma unroll
  for (int e = 0; e < 16; ++e) ax.u[e] = 0;
  if (half == 0) {
    const float* xp = x + bg * 8;
    #pragma unroll
    for (int i = 0; i < 8; ++i) ax.b[i] = f2bf(xp[i]);
  }

  const float dk0 = 1.f - 0.1f * fast_rcp(btau[bl]);
  const float dk1 = 1.f - 0.1f * fast_rcp(btau[16 + bl]);

  float nqp[16], cpl[16];

  for (int s = 0; s < 16; ++s) {
    FragAB ah;
    ah.u = *(const v16u*)&hqA[(s * 32 + lane) << 4];
    const float c0 = corrL[s],      c1 = corrL[16 + s];
    const float c2 = corrL[32 + s], c3 = corrL[48 + s];
    float nqc[16];
    #pragma unroll
    for (int t = 0; t < 2; ++t) {
      FragAB br, bq;
      br.u = *(const v16u*)&wsRec[(((s * 2 + t) * 32 + lane)) << 4];
      bq.u = *(const v16u*)&wsQw [(((s * 2 + t) * 32 + lane)) << 4];
      v8f acc = {0.f, 0.f, 0.f, 0.f, 0.f, 0.f, 0.f, 0.f};
      // rec = h_q(bf16) x recW ; then += x(bf16, K-padded) x qwW
      acc = __builtin_amdgcn_wmma_f32_16x16x32_bf16(false, ah.b, false, br.b,
                                                    (short)0, acc, false, false);
      acc = __builtin_amdgcn_wmma_f32_16x16x32_bf16(false, ax.b, false, bq.b,
                                                    (short)0, acc, false, false);
      const int   k  = (t << 4) + bl;       // C/D layout: N = lane&15
      const float dk = t ? dk1 : dk0;
      #pragma unroll
      for (int r = 0; r < 8; ++r) {
        const int  blocal = r + (half << 3); // C/D layout: M = r + 8*(lane>>4)
        const long bgr    = b0 + blocal;
        const float hval  = hq[((bgr * 32 + k) << 4) + s];
        v4f m = *(const v4f*)&msyn[(blocal * 32 + k) << 2];
        const float hqv = hval - (m.x * c0 + m.y * c1 + m.z * c2 + m.w * c3);
        const float act = fast_tanh(acc[r]);          // V_TANH_F32
        nqc[(t << 3) + r] = hqv * dk + 0.1f * act;
      }
    }
    if (s == 0) {
      #pragma unroll
      for (int j = 0; j < 16; ++j) { nqp[j] = nqc[j]; cpl[j] = 0.f; }
    } else {
      #pragma unroll
      for (int j = 0; j < 16; ++j) {
        const float coup = 0.4f * nqp[j] * nqc[j];           // coupling[s-1]
        const float st   = nqp[j] + 0.1f * (cpl[j] + coup);  // stable[s-1]
        const int k      = ((j >> 3) << 4) + bl;
        const int blocal = (j & 7) + (half << 3);
        stab[((blocal * 32 + k) << 4) + (s - 1)] = st;
        cpl[j] = coup;
        nqp[j] = nqc[j];
      }
    }
  }
  #pragma unroll
  for (int j = 0; j < 16; ++j) {                             // stable[15]
    const float st   = nqp[j] + 0.1f * cpl[j];
    const int k      = ((j >> 3) << 4) + bl;
    const int blocal = (j & 7) + (half << 3);
    stab[((blocal * 32 + k) << 4) + 15] = st;
  }
  __syncthreads();

  // ---- Phase 3: per-(b,s) norms, probs, outputs.
  // Lane owns b = bl, k-rows half*16 .. half*16+15; halves combined via shfl_xor(16).
  float ssq[16], sab[16];
  #pragma unroll
  for (int s = 0; s < 16; ++s) { ssq[s] = 0.f; sab[s] = 0.f; }
  const int kb = half << 4;
  for (int kk = 0; kk < 16; ++kk) {
    v16f row = *(const v16f*)&stab[(bl * 32 + kb + kk) << 4];
    #pragma unroll
    for (int s = 0; s < 16; ++s) {
      const float v = row[s];
      ssq[s] += v * v;
      sab[s] += fabsf(v);
    }
  }
  #pragma unroll
  for (int s = 0; s < 16; ++s) {
    ssq[s] += __shfl_xor(ssq[s], 16, 32);
    sab[s] += __shfl_xor(sab[s], 16, 32);
  }
  float fsc[16], g[16];
  float psum = 0.f;
  #pragma unroll
  for (int s = 0; s < 16; ++s) {
    const float nrm = fast_sqrt(ssq[s]);              // V_SQRT_F32
    const float f   = fast_rcp(nrm + 1e-8f);          // V_RCP_F32
    const float en  = ssq[s] * f * f;                 // energy = sum(q_norm^2)
    const float mg  = sab[s] * f;                     // magnitude = sum(|q_norm|)
    const float p   = mg * fast_rcp(en + 1e-6f);
    fsc[s] = f;
    g[s]   = p;
    psum  += p;
  }
  const float ip = fast_rcp(psum);
  #pragma unroll
  for (int s = 0; s < 16; ++s) g[s] *= fsc[s] * ip;   // g = f * probs

  for (int kk = 0; kk < 16; ++kk) {
    const int k = kb + kk;
    v16f row = *(const v16f*)&stab[(bl * 32 + k) << 4];
    float o = 0.f;
    v16f qn;
    #pragma unroll
    for (int s = 0; s < 16; ++s) {
      o    += row[s] * g[s];
      qn[s] = row[s] * fsc[s];
    }
    // Pure stream-out: non-temporal stores keep L2 free for h_quantum reuse.
    __builtin_nontemporal_store(qn, (v16f*)(qnorm + ((bg * 32 + k) << 4)));
    __builtin_nontemporal_store(o, &out[bg * 32 + k]);
  }
}

// ---------------------------------------------------------------------------
extern "C" void kernel_launch(void* const* d_in, const int* in_sizes, int n_in,
                              void* d_out, int out_size, void* d_ws, size_t ws_size,
                              hipStream_t stream) {
  (void)n_in; (void)out_size; (void)ws_size;
  const float* x    = (const float*)d_in[0];  // (B, 8)
  const float* hq   = (const float*)d_in[1];  // (B, 32, 16)
  const float* qw   = (const float*)d_in[2];  // (8, 32, 16)
  const float* rw   = (const float*)d_in[3];  // (32, 32, 16)
  const float* tau  = (const float*)d_in[4];  // (32)
  const float* syn  = (const float*)d_in[5];  // (4, 16)
  const float* corr = (const float*)d_in[6];  // (4, 16)

  const long B = in_sizes[0] / 8;             // 131072
  float* out   = (float*)d_out;               // output (B,32)
  float* qnorm = out + B * 32;                // q_norm (B,32,16)

  __bf16* wsRec = (__bf16*)d_ws;              // 16*2*32*16 bf16 = 32KB
  __bf16* wsQw  = wsRec + 16 * 2 * 32 * 16;   // +32KB

  prep_weights_kernel<<<64, 256, 0, stream>>>(qw, rw, wsRec, wsQw);
  shqc_kernel<<<(int)(B / 16), 32, 0, stream>>>(x, hq, tau, syn, corr,
                                                wsRec, wsQw, out, qnorm);
}
